// SymphoniesDecoder_35536559407286
// MI455X (gfx1250) — matrix operand order repaired and checked by reading, not compiled, not run on hardware
//
#include <hip/hip_runtime.h>
#include <hip/hip_bf16.h>

// ---------------------------------------------------------------------------
// Symphonies decoder for MI455X (gfx1250, wave32, WMMA).
// All GEMMs: bf16 WMMA (v_wmma_f32_16x16x32_bf16), f32 accumulate.
// Round 4: round-half-up fp32->bf16 (add 0x8000) + single v_perm_b32 pack:
// 3 VALU/pair, pushing the GEMM inner loop below matrix-pipe occupancy.
// ---------------------------------------------------------------------------

#define NH_ 8
#define C_ 128
#define DH_ 16
#define XS_ 128
#define YS_ 128
#define ZS_ 16
#define NVOX_ (XS_ * YS_ * ZS_)   // 262144
#define NFOV_ 65536
#define NQ_ 100
#define LFEAT_ 9196               // 46*152 + 23*76 + 12*38

typedef __attribute__((ext_vector_type(8)))  unsigned short u16x8;
typedef __attribute__((ext_vector_type(16))) unsigned short u16x16;
typedef __attribute__((ext_vector_type(16))) __bf16        bf16x16;
typedef __attribute__((ext_vector_type(8)))  float         f32x8;

// fp32->bf16 round-half-up on both words, pack high halves with one
// v_perm_b32:  D = {hi[31:16], lo[31:16]}  (3 VALU total per pair)
__device__ __forceinline__ unsigned pack2bf(float lo, float hi) {
  unsigned ul = __float_as_uint(lo) + 0x8000u;
  unsigned uh = __float_as_uint(hi) + 0x8000u;
  return __builtin_amdgcn_perm(uh, ul, 0x07060302u);
}

__device__ __forceinline__ float gelu_f(float x) {
  const float k0 = 0.7978845608028654f, k1 = 0.044715f;
  return 0.5f * x * (1.f + tanhf(k0 * (x + k1 * x * x * x)));
}

__device__ __forceinline__ bf16x16 load_frag(const unsigned short* p0,
                                             const unsigned short* p1) {
  u16x8 a = *(const u16x8*)p0;
  u16x8 b = *(const u16x8*)p1;
  u16x16 v = __builtin_shufflevector(a, b, 0,1,2,3,4,5,6,7,8,9,10,11,12,13,14,15);
  return __builtin_bit_cast(bf16x16, v);
}

// ------------------------------- GEMM --------------------------------------
// out[M,N] = act(A[M,K] @ W[K,N] + bias[N]); fp32 in/out, bf16 WMMA inside.
// Block: 128x64 tile, 256 threads = 8 waves in 4x2, each wave 32x32 (2x2 WMMA).
static constexpr int BM = 128, BN = 64, BK = 32;
static constexpr int LDA_ = BK + 8;   // ushort stride 40 (80B) -> conflict-free b128 reads
static constexpr int LDB_ = BK + 8;

template <int ACT, bool GUARD>
__global__ __launch_bounds__(256) void gemm_kernel(
    const float* __restrict__ A, const float* __restrict__ W,
    const float* __restrict__ bias, float* __restrict__ out,
    int M, int Nt, int K) {
  __shared__ unsigned short As[2][BM * LDA_];
  __shared__ unsigned short Ws[2][BN * LDB_];
  const int tid  = threadIdx.x;
  const int lane = tid & 31;
  const int wid  = tid >> 5;
  const int wm   = wid & 3, wn = wid >> 2;
  const int m0   = blockIdx.y * BM, n0 = blockIdx.x * BN;
  const int r0   = wm * 32, c0 = wn * 32;

  f32x8 acc00{}, acc01{}, acc10{}, acc11{};

  // A tile: 16 k-pairs x 16 rows per pass, 8 passes (float2 loads, packed stores)
  const int kA = (tid & 15) * 2;
  const int mA = tid >> 4;
  // W tile: 64 cols x even-k, 4 passes
  const int nW = tid & 63;
  const int kW = (tid >> 6) * 2;
  const int gn  = n0 + nW;
  const int gnc = GUARD ? (gn < Nt ? gn : Nt - 1) : gn;
  const bool nok = GUARD ? (gn < Nt) : true;

  float2 areg[8];
  float2 wreg[4];

  auto load_tiles = [&](int kt) {
#pragma unroll
    for (int r = 0; r < 8; ++r) {
      int gm  = m0 + mA + r * 16;
      int gmc = GUARD ? (gm < M ? gm : M - 1) : gm;
      float2 v = *(const float2*)(A + (size_t)gmc * K + kt + kA);
      if (GUARD && gm >= M) { v.x = 0.f; v.y = 0.f; }
      areg[r] = v;
    }
#pragma unroll
    for (int r = 0; r < 4; ++r) {
      int k = kt + kW + r * 8;
      float2 v;
      v.x = W[(size_t)k * Nt + gnc];
      v.y = W[(size_t)(k + 1) * Nt + gnc];
      if (GUARD && !nok) { v.x = 0.f; v.y = 0.f; }
      wreg[r] = v;
    }
  };
  auto store_tiles = [&](int buf) {
#pragma unroll
    for (int r = 0; r < 8; ++r) {
      int m = mA + r * 16;
      *(unsigned*)&As[buf][m * LDA_ + kA] = pack2bf(areg[r].x, areg[r].y);
    }
#pragma unroll
    for (int r = 0; r < 4; ++r) {
      int k = kW + r * 8;
      *(unsigned*)&Ws[buf][nW * LDB_ + k] = pack2bf(wreg[r].x, wreg[r].y);
    }
  };

  load_tiles(0);
  store_tiles(0);
  const int nk = K / BK;
  for (int it = 0; it < nk; ++it) {
    __syncthreads();
    const int cur = it & 1;
    if (it + 1 < nk) load_tiles((it + 1) * BK);  // prefetch next tile into regs

    const unsigned short* Ab = As[cur];
    const unsigned short* Wb = Ws[cur];
    // A fragment (16x32 bf16): lane row = base+(lane&15); K chunks {kb..kb+7, kb+16..kb+23}
    const int ar  = r0 + (lane & 15);
    const int kbA = (lane >> 4) * 8;
    bf16x16 a0 = load_frag(&Ab[ar * LDA_ + kbA],        &Ab[ar * LDA_ + kbA + 16]);
    bf16x16 a1 = load_frag(&Ab[(ar + 16) * LDA_ + kbA], &Ab[(ar + 16) * LDA_ + kbA + 16]);
    // B fragment (32x16 bf16): lane col = base+(lane&15); K chunks {kb..kb+7, kb+16..kb+23}
    const int bc  = c0 + (lane & 15);
    const int kbB = (lane >> 4) * 8;
    bf16x16 b0 = load_frag(&Wb[bc * LDB_ + kbB],        &Wb[bc * LDB_ + kbB + 16]);
    bf16x16 b1 = load_frag(&Wb[(bc + 16) * LDB_ + kbB], &Wb[(bc + 16) * LDB_ + kbB + 16]);

    acc00 = __builtin_amdgcn_wmma_f32_16x16x32_bf16(false, a0, false, b0, (short)0, acc00, false, false);
    acc01 = __builtin_amdgcn_wmma_f32_16x16x32_bf16(false, a0, false, b1, (short)0, acc01, false, false);
    acc10 = __builtin_amdgcn_wmma_f32_16x16x32_bf16(false, a1, false, b0, (short)0, acc10, false, false);
    acc11 = __builtin_amdgcn_wmma_f32_16x16x32_bf16(false, a1, false, b1, (short)0, acc11, false, false);

    if (it + 1 < nk) store_tiles(cur ^ 1);       // fill the other buffer
  }

  // C/D layout: VGPR i -> row = i + (lane>=16 ? 8 : 0); col = lane & 15
  const int rsel = (lane >> 4) << 3;
  const int csel = lane & 15;
#pragma unroll
  for (int i = 0; i < 8; ++i) {
    const f32x8* accs[4] = {&acc00, &acc01, &acc10, &acc11};
    const int rb[4] = {r0, r0, r0 + 16, r0 + 16};
    const int cb[4] = {c0, c0 + 16, c0, c0 + 16};
#pragma unroll
    for (int tile = 0; tile < 4; ++tile) {
      int gr = m0 + rb[tile] + rsel + i;
      int gc = n0 + cb[tile] + csel;
      if (!GUARD || (gr < M && gc < Nt)) {
        float v = (*accs[tile])[i] + bias[gc];
        if (ACT == 1) v = gelu_f(v);
        out[(size_t)gr * Nt + gc] = v;
      }
    }
  }
}

// ----------------------- transpose / gather / scatter ----------------------
// dst[i*128+c] = src[c*ldsrc + (idx ? idx[i] : i)]   (C-major -> row-major, optional gather)
__global__ __launch_bounds__(256) void xpose_gather_kernel(
    float* __restrict__ dst, const float* __restrict__ src,
    const int* __restrict__ idx, int ldsrc, int ni) {
  __shared__ float tile[32][C_ + 1];
  const int i0 = blockIdx.x * 32;
  const int tx = threadIdx.x, ty = threadIdx.y;
  const int i = i0 + tx;
  const int col = (i < ni) ? (idx ? idx[i] : i) : 0;
#pragma unroll
  for (int cc = 0; cc < 16; ++cc) {
    int c = cc * 8 + ty;
    tile[tx][c] = (i < ni) ? src[(size_t)c * ldsrc + col] : 0.f;
  }
  __syncthreads();
  const int t = ty * 32 + tx;
#pragma unroll
  for (int rch = 0; rch < 16; ++rch) {
    int e = rch * 256 + t;
    int r = e >> 7, c = e & 127;
    if (i0 + r < ni) dst[(size_t)(i0 + r) * C_ + c] = tile[r][c];
  }
}

// dst[c*N + n] = src[n*128 + c]
__global__ __launch_bounds__(256) void xpose_back_kernel(
    float* __restrict__ dst, const float* __restrict__ src, int nrows) {
  __shared__ float tile[32][C_ + 1];
  const int n0 = blockIdx.x * 32;
  const int tx = threadIdx.x, ty = threadIdx.y;
  const int t = ty * 32 + tx;
#pragma unroll
  for (int rch = 0; rch < 16; ++rch) {
    int e = rch * 256 + t;
    int r = e >> 7, c = e & 127;
    if (n0 + r < nrows) tile[r][c] = src[(size_t)(n0 + r) * C_ + c];
  }
  __syncthreads();
#pragma unroll
  for (int cc = 0; cc < 16; ++cc) {
    int c = cc * 8 + ty;
    if (n0 + tx < nrows) dst[(size_t)c * nrows + n0 + tx] = tile[tx][c];
  }
}

__global__ void add_kernel(const float* __restrict__ a, const float* __restrict__ b,
                           float* __restrict__ o, long n) {
  long i = (long)blockIdx.x * 256 + threadIdx.x;
  if (i < n) o[i] = a[i] + b[i];
}

__global__ void scatter_kernel(float* __restrict__ dst, const float* __restrict__ src,
                               const int* __restrict__ idx, int rows) {
  long t = (long)blockIdx.x * 256 + threadIdx.x;
  if (t >= (long)rows * C_) return;
  int i = (int)(t >> 7), c = (int)(t & 127);
  dst[(size_t)idx[i] * C_ + c] = src[t];
}

// ---------------------------- layernorm ------------------------------------
// out = LN(a + b) * g + beta ; one wave per row of 128
__global__ __launch_bounds__(256) void ln_kernel(
    const float* __restrict__ a, const float* __restrict__ b,
    const float* __restrict__ g, const float* __restrict__ beta,
    float* __restrict__ out, int M) {
  int row = blockIdx.x * 8 + threadIdx.y;
  if (row >= M) return;
  int lane = threadIdx.x;
  float4 x = ((const float4*)(a + (size_t)row * C_))[lane];
  float4 rb = ((const float4*)(b + (size_t)row * C_))[lane];
  x.x += rb.x; x.y += rb.y; x.z += rb.z; x.w += rb.w;
  float s  = x.x + x.y + x.z + x.w;
  float sq = x.x * x.x + x.y * x.y + x.z * x.z + x.w * x.w;
#pragma unroll
  for (int o = 16; o > 0; o >>= 1) {
    s  += __shfl_xor(s, o, 32);
    sq += __shfl_xor(sq, o, 32);
  }
  float mean = s * (1.f / C_);
  float var  = sq * (1.f / C_) - mean * mean;
  float rs   = rsqrtf(var + 1e-5f);
  float4 gv = ((const float4*)g)[lane];
  float4 bv = ((const float4*)beta)[lane];
  float4 o4;
  o4.x = (x.x - mean) * rs * gv.x + bv.x;
  o4.y = (x.y - mean) * rs * gv.y + bv.y;
  o4.z = (x.z - mean) * rs * gv.z + bv.z;
  o4.w = (x.w - mean) * rs * gv.w + bv.w;
  ((float4*)(out + (size_t)row * C_))[lane] = o4;
}

// ---------------------------- attention ------------------------------------
// one thread per (query, head); online softmax over NK keys, dh = 16
__global__ __launch_bounds__(256) void attn_kernel(
    const float* __restrict__ Q, const float* __restrict__ K,
    const float* __restrict__ V, float* __restrict__ out, int MQ, int NK) {
  int t = blockIdx.x * 256 + threadIdx.x;
  if (t >= MQ * NH_) return;
  int q = t >> 3, h = t & 7;
  const float* qp = Q + (size_t)q * C_ + h * DH_;
  float qv[DH_], acc[DH_];
#pragma unroll
  for (int d = 0; d < DH_; ++d) { qv[d] = qp[d]; acc[d] = 0.f; }
  float m = -1e30f, l = 0.f;
  for (int k = 0; k < NK; ++k) {
    const float* kp = K + (size_t)k * C_ + h * DH_;
    float s = 0.f;
#pragma unroll
    for (int d = 0; d < DH_; ++d) s += qv[d] * kp[d];
    s *= 0.25f;  // 1/sqrt(16)
    float mn = fmaxf(m, s);
    float so = __expf(m - mn), w = __expf(s - mn);
    l = l * so + w;
    const float* vp = V + (size_t)k * C_ + h * DH_;
#pragma unroll
    for (int d = 0; d < DH_; ++d) acc[d] = acc[d] * so + w * vp[d];
    m = mn;
  }
  float inv = 1.f / l;
  float* op = out + (size_t)q * C_ + h * DH_;
#pragma unroll
  for (int d = 0; d < DH_; ++d) op[d] = acc[d] * inv;
}

// ------------------------- deformable samplers -----------------------------
__global__ __launch_bounds__(256) void sample2d_kernel(
    const float* __restrict__ v, const float* __restrict__ off,
    const float* __restrict__ aw, const float* __restrict__ ref2d,
    float* __restrict__ out) {
  int t = blockIdx.x * 256 + threadIdx.x;
  if (t >= NQ_ * NH_) return;
  int q = t >> 3, h = t & 7;
  float lg[12];
  float mx = -1e30f;
#pragma unroll
  for (int j = 0; j < 12; ++j) { lg[j] = aw[(size_t)q * 96 + h * 12 + j]; mx = fmaxf(mx, lg[j]); }
  float ssum = 0.f;
#pragma unroll
  for (int j = 0; j < 12; ++j) { lg[j] = __expf(lg[j] - mx); ssum += lg[j]; }
  float inv = 1.f / ssum;
  float acc[DH_];
#pragma unroll
  for (int d = 0; d < DH_; ++d) acc[d] = 0.f;
  const int Hs[3] = {46, 23, 12}, Wsz[3] = {152, 76, 38};
  int s = 0;
  for (int l = 0; l < 3; ++l) {
    float rx = ref2d[q * 6 + l * 2 + 0];
    float ry = ref2d[q * 6 + l * 2 + 1];
    int Wd = Wsz[l], Hd = Hs[l];
    for (int p = 0; p < 4; ++p) {
      const float* op = off + (size_t)q * 192 + h * 24 + l * 8 + p * 2;
      float x = rx * Wd + op[0] - 0.5f;
      float y = ry * Hd + op[1] - 0.5f;
      float a = lg[l * 4 + p] * inv;
      float x0 = floorf(x), y0 = floorf(y);
      for (int dx = 0; dx < 2; ++dx)
        for (int dy = 0; dy < 2; ++dy) {
          float xi = x0 + dx, yi = y0 + dy;
          float w = (1.f - fabsf(x - xi)) * (1.f - fabsf(y - yi));
          if (xi >= 0.f && xi <= Wd - 1 && yi >= 0.f && yi <= Hd - 1 && w != 0.f) {
            const float* g = v + ((size_t)(s + (int)yi * Wd + (int)xi)) * C_ + h * DH_;
            float wa = a * w;
#pragma unroll
            for (int d = 0; d < DH_; ++d) acc[d] += wa * g[d];
          }
        }
    }
    s += Hd * Wd;
  }
  float* po = out + (size_t)q * C_ + h * DH_;
#pragma unroll
  for (int d = 0; d < DH_; ++d) po[d] = acc[d];
}

__global__ __launch_bounds__(256) void sample3d_kernel(
    const float* __restrict__ v, const float* __restrict__ off,
    const float* __restrict__ aw, const float* __restrict__ ref,
    const int* __restrict__ idx, float* __restrict__ out, int NQr) {
  int t = blockIdx.x * 256 + threadIdx.x;
  if (t >= NQr * NH_) return;
  int q = t >> 3, h = t & 7;
  int n = idx ? idx[q] : q;
  float rx = ref[(size_t)n * 3 + 0], ry = ref[(size_t)n * 3 + 1], rz = ref[(size_t)n * 3 + 2];
  float lg[8];
  float mx = -1e30f;
#pragma unroll
  for (int p = 0; p < 8; ++p) { lg[p] = aw[(size_t)q * 64 + h * 8 + p]; mx = fmaxf(mx, lg[p]); }
  float ssum = 0.f;
#pragma unroll
  for (int p = 0; p < 8; ++p) { lg[p] = __expf(lg[p] - mx); ssum += lg[p]; }
  float inv = 1.f / ssum;
  float acc[DH_];
#pragma unroll
  for (int d = 0; d < DH_; ++d) acc[d] = 0.f;
  for (int p = 0; p < 8; ++p) {
    const float* op = off + (size_t)q * 192 + h * 24 + p * 3;
    float x = rx * (float)XS_ + op[0] - 0.5f;
    float y = ry * (float)YS_ + op[1] - 0.5f;
    float z = rz * (float)ZS_ + op[2] - 0.5f;
    float a = lg[p] * inv;
    float x0 = floorf(x), y0 = floorf(y), z0 = floorf(z);
    for (int dx = 0; dx < 2; ++dx)
      for (int dy = 0; dy < 2; ++dy)
        for (int dz = 0; dz < 2; ++dz) {
          float xi = x0 + dx, yi = y0 + dy, zi = z0 + dz;
          float w = (1.f - fabsf(x - xi)) * (1.f - fabsf(y - yi)) * (1.f - fabsf(z - zi));
          if (xi >= 0.f && xi <= XS_ - 1 && yi >= 0.f && yi <= YS_ - 1 &&
              zi >= 0.f && zi <= ZS_ - 1 && w != 0.f) {
            size_t vi = ((size_t)((int)xi * YS_ + (int)yi) * ZS_ + (int)zi);
            const float* g = v + vi * C_ + h * DH_;
            float wa = a * w;
#pragma unroll
            for (int d = 0; d < DH_; ++d) acc[d] += wa * g[d];
          }
        }
  }
  float* po = out + (size_t)q * C_ + h * DH_;
#pragma unroll
  for (int d = 0; d < DH_; ++d) po[d] = acc[d];
}

// ------------------------------ host side ----------------------------------
static inline void gemm(hipStream_t s, const float* A, const float* W, const float* b,
                        float* out, int M, int N, int K, bool act_gelu) {
  dim3 g((N + BN - 1) / BN, (M + BM - 1) / BM);
  const bool guard = (M % BM) != 0 || (N % BN) != 0;
  if (act_gelu) {
    if (guard) gemm_kernel<1, true ><<<g, 256, 0, s>>>(A, W, b, out, M, N, K);
    else       gemm_kernel<1, false><<<g, 256, 0, s>>>(A, W, b, out, M, N, K);
  } else {
    if (guard) gemm_kernel<0, true ><<<g, 256, 0, s>>>(A, W, b, out, M, N, K);
    else       gemm_kernel<0, false><<<g, 256, 0, s>>>(A, W, b, out, M, N, K);
  }
}
static inline void lnorm(hipStream_t s, const float* a, const float* b, const float* g,
                         const float* beta, float* out, int M) {
  ln_kernel<<<dim3((M + 7) / 8), dim3(32, 8), 0, s>>>(a, b, g, beta, out, M);
}
static inline void vadd(hipStream_t s, const float* a, const float* b, float* o, long n) {
  add_kernel<<<dim3((unsigned)((n + 255) / 256)), 256, 0, s>>>(a, b, o, n);
}

extern "C" void kernel_launch(void* const* d_in, const int* in_sizes, int n_in,
                              void* d_out, int out_size, void* d_ws, size_t ws_size,
                              hipStream_t stream) {
  (void)in_sizes; (void)n_in; (void)out_size; (void)ws_size;
  const float* scene_embed  = (const float*)d_in[0];
  const float* inst_queries = (const float*)d_in[1];
  const float* feat0 = (const float*)d_in[2];
  const float* feat1 = (const float*)d_in[3];
  const float* feat2 = (const float*)d_in[4];
  const float* scene_pos = (const float*)d_in[5];
  const float* inst_pos  = (const float*)d_in[6];
  const float* ref_2d = (const float*)d_in[7];
  const float* ref_3d = (const float*)d_in[8];
  const float* ref_vox = (const float*)d_in[9];
  const int*   fov_idx = (const int*)d_in[10];
  auto Pp = [&](int i) { return (const float*)d_in[i]; };
  // params: jax tree-flatten order (sorted dict keys, depth first)
  const float *l1_aw_b=Pp(11),*l1_aw_w=Pp(12),*l1_o_b=Pp(13),*l1_o_w=Pp(14),
              *l1_off_b=Pp(15),*l1_off_w=Pp(16),*l1_v_b=Pp(17),*l1_v_w=Pp(18),
              *l1_fc1_b=Pp(19),*l1_fc1_w=Pp(20),*l1_fc2_b=Pp(21),*l1_fc2_w=Pp(22),
              *l1_ln1_b=Pp(23),*l1_ln1_g=Pp(24),*l1_ln2_b=Pp(25),*l1_ln2_g=Pp(26);
  const float *l2_k_b=Pp(27),*l2_k_w=Pp(28),*l2_o_b=Pp(29),*l2_o_w=Pp(30),
              *l2_q_b=Pp(31),*l2_q_w=Pp(32),*l2_v_b=Pp(33),*l2_v_w=Pp(34),
              *l2_ln1_b=Pp(35),*l2_ln1_g=Pp(36);
  const float *l3_aw_b=Pp(37),*l3_aw_w=Pp(38),*l3_o_b=Pp(39),*l3_o_w=Pp(40),
              *l3_off_b=Pp(41),*l3_off_w=Pp(42),*l3_v_b=Pp(43),*l3_v_w=Pp(44),
              *l3_fc1_b=Pp(45),*l3_fc1_w=Pp(46),*l3_fc2_b=Pp(47),*l3_fc2_w=Pp(48),
              *l3_ln1_b=Pp(49),*l3_ln1_g=Pp(50),*l3_ln2_b=Pp(51),*l3_ln2_g=Pp(52);
  const float *l4_aw_b=Pp(53),*l4_aw_w=Pp(54),*l4_o_b=Pp(55),*l4_o_w=Pp(56),
              *l4_off_b=Pp(57),*l4_off_w=Pp(58),*l4_v_b=Pp(59),*l4_v_w=Pp(60),
              *l4_ln1_b=Pp(61),*l4_ln1_g=Pp(62);
  const float *l5_k_b=Pp(63),*l5_k_w=Pp(64),*l5_o_b=Pp(65),*l5_o_w=Pp(66),
              *l5_q_b=Pp(67),*l5_q_w=Pp(68),*l5_v_b=Pp(69),*l5_v_w=Pp(70),
              *l5_fc1_b=Pp(71),*l5_fc1_w=Pp(72),*l5_fc2_b=Pp(73),*l5_fc2_w=Pp(74),
              *l5_ln1_b=Pp(75),*l5_ln1_g=Pp(76),*l5_ln2_b=Pp(77),*l5_ln2_g=Pp(78);

  // workspace carve-up (~508 MB of floats)
  float* ws = (float*)d_ws;
  size_t o = 0;
  float* scene_flat = ws + o; o += (size_t)NVOX_ * C_;   // persistent (N, C)
  float* vbuf = ws + o; o += (size_t)NVOX_ * C_;         // v1 / v3 / ffn-hidden / v4
  float* F0 = ws + o; o += (size_t)NFOV_ * C_;           // fov state
  float* PF = ws + o; o += (size_t)NFOV_ * C_;           // pos_fov
  float* T1 = ws + o; o += (size_t)NFOV_ * C_;
  float* T2 = ws + o; o += (size_t)NFOV_ * C_;
  float* T3 = ws + o; o += (size_t)NFOV_ * C_;
  float* offb = ws + o; o += (size_t)NFOV_ * 192;
  float* awb  = ws + o; o += (size_t)NFOV_ * 64;
  float* feats = ws + o; o += (size_t)LFEAT_ * C_;
  float* sinst = ws + o; o += NQ_ * C_;
  float* si1 = ws + o; o += NQ_ * C_;
  float* st1 = ws + o; o += NQ_ * C_;
  float* st2 = ws + o; o += NQ_ * C_;
  float* sq = ws + o; o += NQ_ * C_;
  float* sk = ws + o; o += NQ_ * C_;
  float* sv = ws + o; o += NQ_ * C_;
  float* sh = ws + o; o += NQ_ * 512;
  float* soff = ws + o; o += NQ_ * 192;
  float* saw = ws + o; o += NQ_ * 96;
  float* ssamp = ws + o; o += NQ_ * C_;

  const dim3 tb(32, 8);
  // ---- Phase A: layout prep -------------------------------------------------
  xpose_gather_kernel<<<dim3(NVOX_ / 32), tb, 0, stream>>>(scene_flat, scene_embed, nullptr, NVOX_, NVOX_);
  xpose_gather_kernel<<<dim3(NFOV_ / 32), tb, 0, stream>>>(F0, scene_embed, fov_idx, NVOX_, NFOV_);
  xpose_gather_kernel<<<dim3(NFOV_ / 32), tb, 0, stream>>>(PF, scene_pos, fov_idx, NVOX_, NFOV_);
  xpose_gather_kernel<<<dim3((6992 + 31) / 32), tb, 0, stream>>>(feats, feat0, nullptr, 6992, 6992);
  xpose_gather_kernel<<<dim3((1748 + 31) / 32), tb, 0, stream>>>(feats + (size_t)6992 * C_, feat1, nullptr, 1748, 1748);
  xpose_gather_kernel<<<dim3((456 + 31) / 32), tb, 0, stream>>>(feats + (size_t)8740 * C_, feat2, nullptr, 456, 456);

  // ---- Phase B: l1 deform2d on instance queries ----------------------------
  vadd(stream, inst_queries, inst_pos, si1, (long)NQ_ * C_);
  gemm(stream, feats, l1_v_w, l1_v_b, vbuf, LFEAT_, C_, C_, false);
  gemm(stream, si1, l1_off_w, l1_off_b, soff, NQ_, 192, C_, false);
  gemm(stream, si1, l1_aw_w, l1_aw_b, saw, NQ_, 96, C_, false);
  sample2d_kernel<<<dim3((NQ_ * NH_ + 255) / 256), 256, 0, stream>>>(vbuf, soff, saw, ref_2d, ssamp);
  gemm(stream, ssamp, l1_o_w, l1_o_b, st1, NQ_, C_, C_, false);
  lnorm(stream, inst_queries, st1, l1_ln1_g, l1_ln1_b, st2, NQ_);
  gemm(stream, st2, l1_fc1_w, l1_fc1_b, sh, NQ_, 512, C_, true);
  gemm(stream, sh, l1_fc2_w, l1_fc2_b, st1, NQ_, C_, 512, false);
  lnorm(stream, st2, st1, l1_ln2_g, l1_ln2_b, sinst, NQ_);

  // ---- Phase C: l2 cross-attention fov -> inst -----------------------------
  vadd(stream, sinst, inst_pos, si1, (long)NQ_ * C_);
  gemm(stream, si1, l2_k_w, l2_k_b, sk, NQ_, C_, C_, false);
  gemm(stream, sinst, l2_v_w, l2_v_b, sv, NQ_, C_, C_, false);
  vadd(stream, F0, PF, T1, (long)NFOV_ * C_);
  gemm(stream, T1, l2_q_w, l2_q_b, T2, NFOV_, C_, C_, false);
  attn_kernel<<<dim3(NFOV_ * NH_ / 256), 256, 0, stream>>>(T2, sk, sv, T3, NFOV_, NQ_);
  gemm(stream, T3, l2_o_w, l2_o_b, T1, NFOV_, C_, C_, false);
  lnorm(stream, F0, T1, l2_ln1_g, l2_ln1_b, F0, NFOV_);

  // ---- Phase D: l3 deform3d on fov over scene ------------------------------
  gemm(stream, scene_flat, l3_v_w, l3_v_b, vbuf, NVOX_, C_, C_, false);
  vadd(stream, F0, PF, T1, (long)NFOV_ * C_);
  gemm(stream, T1, l3_off_w, l3_off_b, offb, NFOV_, 192, C_, false);
  gemm(stream, T1, l3_aw_w, l3_aw_b, awb, NFOV_, 64, C_, false);
  sample3d_kernel<<<dim3(NFOV_ * NH_ / 256), 256, 0, stream>>>(vbuf, offb, awb, ref_vox, fov_idx, T2, NFOV_);
  gemm(stream, T2, l3_o_w, l3_o_b, T3, NFOV_, C_, C_, false);
  lnorm(stream, F0, T3, l3_ln1_g, l3_ln1_b, F0, NFOV_);
  gemm(stream, F0, l3_fc1_w, l3_fc1_b, vbuf, NFOV_, 512, C_, true);  // hidden reuses vbuf
  gemm(stream, vbuf, l3_fc2_w, l3_fc2_b, T1, NFOV_, C_, 512, false);
  lnorm(stream, F0, T1, l3_ln2_g, l3_ln2_b, F0, NFOV_);
  scatter_kernel<<<dim3((unsigned)((long)NFOV_ * C_ / 256)), 256, 0, stream>>>(scene_flat, F0, fov_idx, NFOV_);

  // ---- Phase E: l4 deform3d on inst over updated scene ---------------------
  gemm(stream, scene_flat, l4_v_w, l4_v_b, vbuf, NVOX_, C_, C_, false);
  vadd(stream, sinst, inst_pos, si1, (long)NQ_ * C_);
  gemm(stream, si1, l4_off_w, l4_off_b, soff, NQ_, 192, C_, false);
  gemm(stream, si1, l4_aw_w, l4_aw_b, saw, NQ_, 64, C_, false);
  sample3d_kernel<<<dim3((NQ_ * NH_ + 255) / 256), 256, 0, stream>>>(vbuf, soff, saw, ref_3d, nullptr, ssamp, NQ_);
  gemm(stream, ssamp, l4_o_w, l4_o_b, st1, NQ_, C_, C_, false);
  lnorm(stream, sinst, st1, l4_ln1_g, l4_ln1_b, sinst, NQ_);

  // ---- Phase F: l5 self-attention + FFN on inst ----------------------------
  float* out_f = (float*)d_out;
  vadd(stream, sinst, inst_pos, si1, (long)NQ_ * C_);
  gemm(stream, si1, l5_q_w, l5_q_b, sq, NQ_, C_, C_, false);
  gemm(stream, si1, l5_k_w, l5_k_b, sk, NQ_, C_, C_, false);
  gemm(stream, sinst, l5_v_w, l5_v_b, sv, NQ_, C_, C_, false);
  attn_kernel<<<dim3((NQ_ * NH_ + 255) / 256), 256, 0, stream>>>(sq, sk, sv, st1, NQ_, NQ_);
  gemm(stream, st1, l5_o_w, l5_o_b, st2, NQ_, C_, C_, false);
  lnorm(stream, sinst, st2, l5_ln1_g, l5_ln1_b, si1, NQ_);
  gemm(stream, si1, l5_fc1_w, l5_fc1_b, sh, NQ_, 512, C_, true);
  gemm(stream, sh, l5_fc2_w, l5_fc2_b, st1, NQ_, C_, 512, false);
  lnorm(stream, si1, st1, l5_ln2_g, l5_ln2_b, out_f + (size_t)NVOX_ * C_, NQ_);

  // ---- Phase G: scene output (N,C) -> (C,N) --------------------------------
  xpose_back_kernel<<<dim3(NVOX_ / 32), tb, 0, stream>>>(out_f, scene_flat, NVOX_);
}
